// model_rel_gru_mask_57861799411873
// MI455X (gfx1250) — matrix operand order repaired and checked by reading, not compiled, not run on hardware
//
#include <hip/hip_runtime.h>
#include <hip/hip_bf16.h>
#include <cstddef>
#include <cstdint>

typedef _Float16 half_t;
typedef __attribute__((ext_vector_type(16))) _Float16 v16h;
typedef __attribute__((ext_vector_type(8)))  float    v8f;

#define S_SCENES 1024
#define L_AG     20
#define T_PAST   8
#define F_FUT    12
#define C_DIM    256
#define M_DIM    256
#define NH       4
#define SPACE_   5
#define N_SLOT   (SPACE_ * L_AG)          // 100
#define A_TOT    (S_SCENES * L_AG)        // 20480
#define D_DIM    (C_DIM + M_DIM)          // 512
#define R_DEC    (A_TOT * NH)             // 81920
#define EPS_     1e-8f

// ---------------------------------------------------------------------------
// NT GEMM on the CDNA5 WMMA path:  Y[M,N_] = X[M,K_] (f32) * W[N_,K_]^T (f16)
// Compile-time N_/K_ so every bounds check folds; branch-free inner loop.
// Wave tile: 16(M) x 64(N).  4 waves / block.
//
// A fragment (16x32 f16, wave32): lane m=lane&15, hsel=lane>>4:
//   elem i -> K = k0 + hsel*16 + i
// B fragment (32x16 f16): lane n=lane&15, kw = k0 + hsel*16:
//   elem i -> W[n, kw+i]  (single aligned 32-byte v16h load)
// C/D (16x16 f32): elem v -> row m0 + v + 8*hsel, col = lane&15.
// ---------------------------------------------------------------------------
template <int N_, int K_>
__global__ __launch_bounds__(128) void gemm_nt(
    const float* __restrict__ X, const half_t* __restrict__ W,
    float* __restrict__ Y, int Mrows)
{
    static_assert(K_ % 32 == 0 || K_ % 32 == 16, "K tail must be 0 or 16");
    constexpr int KFULL = (K_ / 32) * 32;

    const int lane = threadIdx.x & 31;
    const int wave = threadIdx.x >> 5;
    const int m0   = (blockIdx.x * 4 + wave) * 16;
    const int n0   = blockIdx.y * 64;
    const int mrow = m0 + (lane & 15);
    const int hsel = lane >> 4;
    const int bn   = n0 + (lane & 15);

    v8f zero = {};
    v8f acc[4];
#pragma unroll
    for (int t = 0; t < 4; ++t) acc[t] = zero;

    const float* xr = X + (size_t)mrow * K_;

    // Clamp out-of-range B rows to row 0 (results masked at store time).
    const half_t* wrow[4];
#pragma unroll
    for (int t = 0; t < 4; ++t) {
        const int n = bn + t * 16;
        wrow[t] = W + (size_t)((n < N_) ? n : 0) * K_;
    }

    // ---- full 32-wide K tiles: completely unguarded, vectorized ----
    for (int k0 = 0; k0 < KFULL; k0 += 32) {
        const int ka = k0 + hsel * 16;
        const int kb = k0 + hsel * 16 + 8;
        float4 lo0 = *(const float4*)(xr + ka);
        float4 lo1 = *(const float4*)(xr + ka + 4);
        float4 hi0 = *(const float4*)(xr + kb);
        float4 hi1 = *(const float4*)(xr + kb + 4);
        v16h af;
        af[0]  = (half_t)lo0.x; af[1]  = (half_t)lo0.y;
        af[2]  = (half_t)lo0.z; af[3]  = (half_t)lo0.w;
        af[4]  = (half_t)lo1.x; af[5]  = (half_t)lo1.y;
        af[6]  = (half_t)lo1.z; af[7]  = (half_t)lo1.w;
        af[8]  = (half_t)hi0.x; af[9]  = (half_t)hi0.y;
        af[10] = (half_t)hi0.z; af[11] = (half_t)hi0.w;
        af[12] = (half_t)hi1.x; af[13] = (half_t)hi1.y;
        af[14] = (half_t)hi1.z; af[15] = (half_t)hi1.w;

        const int kw = k0 + hsel * 16;
#pragma unroll
        for (int t = 0; t < 4; ++t) {
            v16h bf = *(const v16h*)(wrow[t] + kw);
            acc[t] = __builtin_amdgcn_wmma_f32_16x16x32_f16(
                false, af, false, bf, (short)0, acc[t], false, false);
        }
    }

    // ---- compile-time tail tile (K_ % 32 == 16) ----
    if (KFULL < K_) {
        // hsel==0: elem i -> K = KFULL + i < K_ (valid); hsel==1: K >= K_ -> zero
        v16h af = {};
        if (hsel == 0) {
            float4 lo0 = *(const float4*)(xr + KFULL);
            float4 lo1 = *(const float4*)(xr + KFULL + 4);
            float4 hi0 = *(const float4*)(xr + KFULL + 8);
            float4 hi1 = *(const float4*)(xr + KFULL + 12);
            af[0]  = (half_t)lo0.x; af[1]  = (half_t)lo0.y;
            af[2]  = (half_t)lo0.z; af[3]  = (half_t)lo0.w;
            af[4]  = (half_t)lo1.x; af[5]  = (half_t)lo1.y;
            af[6]  = (half_t)lo1.z; af[7]  = (half_t)lo1.w;
            af[8]  = (half_t)hi0.x; af[9]  = (half_t)hi0.y;
            af[10] = (half_t)hi0.z; af[11] = (half_t)hi0.w;
            af[12] = (half_t)hi1.x; af[13] = (half_t)hi1.y;
            af[14] = (half_t)hi1.z; af[15] = (half_t)hi1.w;
        }

#pragma unroll
        for (int t = 0; t < 4; ++t) {
            // B valid only for hsel==0 (kw = KFULL..KFULL+15); hsel==1 is past K_
            v16h bf = {};
            if (hsel == 0) bf = *(const v16h*)(wrow[t] + KFULL);
            acc[t] = __builtin_amdgcn_wmma_f32_16x16x32_f16(
                false, af, false, bf, (short)0, acc[t], false, false);
        }
    }

    // ---- store C/D ----
#pragma unroll
    for (int t = 0; t < 4; ++t) {
        const int col = bn + t * 16;
        if (col < N_) {
#pragma unroll
            for (int v = 0; v < 8; ++v) {
                const int row = m0 + v + hsel * 8;
                Y[(size_t)row * N_ + col] = acc[t][v];
            }
        }
    }
}

// ---------------------------------------------------------------------------
// GRU gate fusion: h = (1-z)*n + z*h  (biases folded in here, not in GEMM)
// ---------------------------------------------------------------------------
__global__ void gru_gate(const float* __restrict__ gi, const float* __restrict__ gh,
                         const float* __restrict__ bih, const float* __restrict__ bhh,
                         float* __restrict__ h, int R, int H)
{
    int id = blockIdx.x * blockDim.x + threadIdx.x;
    if (id >= R * H) return;
    int row = id / H, j = id % H;
    const float* gir = gi + (size_t)row * 3 * H;
    const float* ghr = gh + (size_t)row * 3 * H;
    float ir  = gir[j]         + bih[j],         hr = ghr[j]         + bhh[j];
    float iz  = gir[H + j]     + bih[H + j],     hz = ghr[H + j]     + bhh[H + j];
    float in_ = gir[2 * H + j] + bih[2 * H + j], hn = ghr[2 * H + j] + bhh[2 * H + j];
    float rg = 1.f / (1.f + __expf(-(ir + hr)));
    float z  = 1.f / (1.f + __expf(-(iz + hz)));
    float n  = tanhf(in_ + rg * hn);
    size_t hi = (size_t)row * H + j;
    h[hi] = (1.f - z) * n + z * h[hi];
}

// ---------------------------------------------------------------------------
// Small helpers
// ---------------------------------------------------------------------------
__global__ void cvt_f16(const float* __restrict__ x, half_t* __restrict__ y, int n)
{
    int i = blockIdx.x * blockDim.x + threadIdx.x;
    if (i < n) y[i] = (half_t)x[i];
}

__global__ void fill_f32(float* __restrict__ p, float v, size_t n)
{
    size_t i = (size_t)blockIdx.x * blockDim.x + threadIdx.x;
    size_t stride = (size_t)gridDim.x * blockDim.x;
    for (; i < n; i += stride) p[i] = v;
}

__global__ void bcast_row(float* __restrict__ dst, const float* __restrict__ src,
                          int R, int H)
{
    int id = blockIdx.x * blockDim.x + threadIdx.x;
    if (id < R * H) dst[id] = src[id % H];
}

__global__ void init_present(float* __restrict__ present, const float* __restrict__ past)
{
    int id = blockIdx.x * blockDim.x + threadIdx.x;
    if (id >= R_DEC * 2) return;
    int row = id >> 1, c = id & 1;
    int a = row / NH;
    present[id] = past[((size_t)a * T_PAST + (T_PAST - 1)) * 2 + c];
}

// emb = relu(x @ w1^T + b1) @ w2^T + b2 ; x:[A*T,2] -> out time-major [T][A,16]
__global__ void embed_mlp(const float* __restrict__ x,
                          const float* __restrict__ w1, const float* __restrict__ b1,
                          const float* __restrict__ w2, const float* __restrict__ b2,
                          float* __restrict__ out)
{
    int rid = blockIdx.x * blockDim.x + threadIdx.x;
    if (rid >= A_TOT * T_PAST) return;
    int a = rid / T_PAST, t = rid % T_PAST;
    float x0 = x[(size_t)rid * 2], x1 = x[(size_t)rid * 2 + 1];
    float hb[8];
#pragma unroll
    for (int i = 0; i < 8; ++i) {
        float v = x0 * w1[i * 2] + x1 * w1[i * 2 + 1] + b1[i];
        hb[i] = v > 0.f ? v : 0.f;
    }
    float* o = out + ((size_t)t * A_TOT + a) * 16;
#pragma unroll
    for (int j = 0; j < 16; ++j) {
        float v = b2[j];
#pragma unroll
        for (int i = 0; i < 8; ++i) v += hb[i] * w2[j * 8 + i];
        o[j] = v;
    }
}

// x_cat[a, 0:16] = emb (or 0), x_cat[a, 16:272] = r
__global__ void concat_emb_r(const float* __restrict__ emb, const float* __restrict__ r,
                             float* __restrict__ xcat, int use_emb)
{
    int id = blockIdx.x * blockDim.x + threadIdx.x;
    if (id >= A_TOT * 272) return;
    int a = id / 272, j = id % 272;
    float v;
    if (j < 16) v = use_emb ? emb[(size_t)a * 16 + j] : 0.f;
    else        v = r[(size_t)a * 256 + (j - 16)];
    xcat[id] = v;
}

__device__ __forceinline__ float block_reduce_256(float* red, int j, float v)
{
    red[j] = v;
    __syncthreads();
#pragma unroll
    for (int s = 128; s > 0; s >>= 1) {
        if (j < s) red[j] += red[j + s];
        __syncthreads();
    }
    float r = red[0];
    __syncthreads();
    return r;
}

// ---------------------------------------------------------------------------
// Memory write: content-addressed write into agent's own 5-slot block.
// One 256-thread block per agent.
// ---------------------------------------------------------------------------
__global__ __launch_bounds__(256) void mem_write(
    const float* __restrict__ wp, const float* __restrict__ wb,
    float* __restrict__ mem)
{
    const int a = blockIdx.x;
    const int s = a / L_AG, l = a % L_AG;
    const int j = threadIdx.x;                // j == m in [0,256)
    __shared__ float red[256];
    __shared__ float sh_dot[SPACE_], sh_nrm[SPACE_], sh_ww[SPACE_];
    __shared__ float sh_knorm, sh_beta;

    const float* row = wp + (size_t)a * (3 * M_DIM + 1);
    float kv = tanhf(row[j] + wb[j]);
    float ev = 1.f / (1.f + __expf(-(row[257 + j] + wb[257 + j])));
    float av = tanhf(row[513 + j] + wb[513 + j]);
    if (j == 0) {
        float bx = row[256] + wb[256];
        sh_beta = (bx > 20.f) ? bx : log1pf(__expf(bx));
    }
    float ks = block_reduce_256(red, j, kv * kv);
    if (j == 0) sh_knorm = sqrtf(ks);

    float* mblk = mem + ((size_t)s * N_SLOT + l * SPACE_) * M_DIM;
    float mv[SPACE_];
#pragma unroll
    for (int p = 0; p < SPACE_; ++p) {
        mv[p] = mblk[p * M_DIM + j];
        float d = block_reduce_256(red, j, kv * mv[p]);
        float q = block_reduce_256(red, j, mv[p] * mv[p]);
        if (j == 0) { sh_dot[p] = d; sh_nrm[p] = sqrtf(q); }
    }
    __syncthreads();
    if (j == 0) {
        float sc[SPACE_], mx = -3.4e38f;
#pragma unroll
        for (int p = 0; p < SPACE_; ++p) {
            sc[p] = sh_beta * sh_dot[p] / ((sh_knorm + EPS_) * (sh_nrm[p] + EPS_));
            mx = fmaxf(mx, sc[p]);
        }
        float sum = 0.f;
#pragma unroll
        for (int p = 0; p < SPACE_; ++p) { sc[p] = __expf(sc[p] - mx); sum += sc[p]; }
        float inv = 1.f / sum;
#pragma unroll
        for (int p = 0; p < SPACE_; ++p) sh_ww[p] = sc[p] * inv;
    }
    __syncthreads();
#pragma unroll
    for (int p = 0; p < SPACE_; ++p) {
        float w = sh_ww[p];
        mblk[p * M_DIM + j] = mv[p] * (1.f - w * ev) + w * av;
    }
}

// ---------------------------------------------------------------------------
// Memory read: 4-head content addressing over 100 slots with self-block mask,
// masked softmax, reads; writes r (mean over heads) and the decoder input
// x_dec[a*NH+h] = [h_enc | read_h].  One 256-thread block per agent.
// ---------------------------------------------------------------------------
__global__ __launch_bounds__(256) void mem_read(
    const float* __restrict__ rp, const float* __restrict__ rb,
    const float* __restrict__ mem, const float* __restrict__ h,
    float* __restrict__ r_out, float* __restrict__ x_dec)
{
    const int a = blockIdx.x;
    const int s = a / L_AG, l = a % L_AG;
    const int j = threadIdx.x;
    __shared__ float kr[NH][M_DIM];
    __shared__ float red[256];
    __shared__ float sc[NH][N_SLOT];
    __shared__ float beta[NH], knrm[NH];

    const float* row = rp + (size_t)a * (NH * (M_DIM + 1));
#pragma unroll
    for (int hh = 0; hh < NH; ++hh)
        kr[hh][j] = tanhf(row[hh * 257 + j] + rb[hh * 257 + j]);
    if (j < NH) {
        float bx = row[j * 257 + 256] + rb[j * 257 + 256];
        beta[j] = (bx > 20.f) ? bx : log1pf(__expf(bx));
    }
    __syncthreads();
#pragma unroll
    for (int hh = 0; hh < NH; ++hh) {
        float kss = block_reduce_256(red, j, kr[hh][j] * kr[hh][j]);
        if (j == 0) knrm[hh] = sqrtf(kss);
    }
    __syncthreads();

    if (j < N_SLOT) {
        const float* mrow = mem + ((size_t)s * N_SLOT + j) * M_DIM;
        float d[NH] = {0.f, 0.f, 0.f, 0.f};
        float nn = 0.f;
        for (int m = 0; m < M_DIM; ++m) {
            float mvv = mrow[m];
            nn += mvv * mvv;
#pragma unroll
            for (int hh = 0; hh < NH; ++hh) d[hh] += kr[hh][m] * mvv;
        }
        float inv = 1.f / (sqrtf(nn) + EPS_);
        bool allowed = (j / SPACE_) != l;
#pragma unroll
        for (int hh = 0; hh < NH; ++hh)
            sc[hh][j] = allowed ? beta[hh] * d[hh] * inv / (knrm[hh] + EPS_) : -1e9f;
    }
    __syncthreads();
    if (j < NH) {
        float mx = -3.4e38f;
        for (int n = 0; n < N_SLOT; ++n) mx = fmaxf(mx, sc[j][n]);
        float sum = 0.f;
        for (int n = 0; n < N_SLOT; ++n) {
            float e = __expf(sc[j][n] - mx);
            sc[j][n] = e; sum += e;
        }
        float inv = 1.f / sum;
        for (int n = 0; n < N_SLOT; ++n) sc[j][n] *= inv;
    }
    __syncthreads();

    float hv = h[(size_t)a * C_DIM + j];
    float racc = 0.f;
#pragma unroll
    for (int hh = 0; hh < NH; ++hh) {
        float rd = 0.f;
        for (int n = 0; n < N_SLOT; ++n)
            rd += sc[hh][n] * mem[((size_t)s * N_SLOT + n) * M_DIM + j];
        racc += rd;
        float* xr = x_dec + (size_t)(a * NH + hh) * D_DIM;
        xr[j] = hv;
        xr[C_DIM + j] = rd;
    }
    r_out[(size_t)a * M_DIM + j] = racc * (1.f / NH);
}

// ---------------------------------------------------------------------------
// out = h_fut @ out_w^T + out_b ; present += out ; d_out[a,h,f,:] = present
// One wave (32 lanes) per decoder row; shfl reduction.
// ---------------------------------------------------------------------------
__global__ __launch_bounds__(256) void dec_out(
    const float* __restrict__ hfut, const float* __restrict__ ow,
    const float* __restrict__ ob, float* __restrict__ present,
    float* __restrict__ out, int f)
{
    int wid = threadIdx.x >> 5, lane = threadIdx.x & 31;
    int row = blockIdx.x * 8 + wid;
    if (row >= R_DEC) return;
    const float* hr = hfut + (size_t)row * D_DIM;
    float s0 = 0.f, s1 = 0.f;
    for (int k = lane; k < D_DIM; k += 32) {
        float hv = hr[k];
        s0 += hv * ow[k];
        s1 += hv * ow[D_DIM + k];
    }
#pragma unroll
    for (int off = 16; off > 0; off >>= 1) {
        s0 += __shfl_down(s0, off, 32);
        s1 += __shfl_down(s1, off, 32);
    }
    if (lane == 0) {
        float p0 = present[(size_t)row * 2]     + s0 + ob[0];
        float p1 = present[(size_t)row * 2 + 1] + s1 + ob[1];
        present[(size_t)row * 2]     = p0;
        present[(size_t)row * 2 + 1] = p1;
        out[((size_t)row * F_FUT + f) * 2]     = p0;
        out[((size_t)row * F_FUT + f) * 2 + 1] = p1;
    }
}

// ---------------------------------------------------------------------------
template <int N_, int K_>
static inline void launch_gemm(const float* X, const half_t* W, float* Y,
                               int M, hipStream_t stream)
{
    dim3 g((unsigned)(M / 64), (unsigned)((N_ + 63) / 64));
    gemm_nt<N_, K_><<<g, dim3(128), 0, stream>>>(X, W, Y, M);
}

extern "C" void kernel_launch(void* const* d_in, const int* in_sizes, int n_in,
                              void* d_out, int out_size, void* d_ws, size_t ws_size,
                              hipStream_t stream)
{
    (void)in_sizes; (void)n_in; (void)out_size; (void)ws_size;
    const float* past     = (const float*)d_in[0];
    const float* past_rel = (const float*)d_in[1];
    const float* ep_w1 = (const float*)d_in[3];
    const float* ep_b1 = (const float*)d_in[4];
    const float* ep_w2 = (const float*)d_in[5];
    const float* ep_b2 = (const float*)d_in[6];
    const float* er_w1 = (const float*)d_in[7];
    const float* er_b1 = (const float*)d_in[8];
    const float* er_w2 = (const float*)d_in[9];
    const float* er_b2 = (const float*)d_in[10];
    const float* enc_wih = (const float*)d_in[11];
    const float* enc_whh = (const float*)d_in[12];
    const float* enc_bih = (const float*)d_in[13];
    const float* enc_bhh = (const float*)d_in[14];
    const float* rel_wih = (const float*)d_in[15];
    const float* rel_whh = (const float*)d_in[16];
    const float* rel_bih = (const float*)d_in[17];
    const float* rel_bhh = (const float*)d_in[18];
    const float* dec_wih = (const float*)d_in[19];
    const float* dec_whh = (const float*)d_in[20];
    const float* dec_bih = (const float*)d_in[21];
    const float* dec_bhh = (const float*)d_in[22];
    const float* out_w   = (const float*)d_in[23];
    const float* out_b   = (const float*)d_in[24];
    const float* read_w  = (const float*)d_in[25];
    const float* read_b  = (const float*)d_in[26];
    const float* write_w = (const float*)d_in[27];
    const float* write_b = (const float*)d_in[28];
    const float* init_r  = (const float*)d_in[29];
    const float* h_bias      = (const float*)d_in[30];
    const float* h_bias_rel  = (const float*)d_in[31];
    const float* h_bias_fut  = (const float*)d_in[32];

    char* base = (char*)d_ws;
    size_t off = 0;
    auto allocH = [&](size_t n) -> half_t* {
        half_t* p = (half_t*)(base + off);
        off += ((n * sizeof(half_t) + 255) / 256) * 256;
        return p;
    };
    auto allocF = [&](size_t n) -> float* {
        float* p = (float*)(base + off);
        off += ((n * sizeof(float) + 255) / 256) * 256;
        return p;
    };

    // fp16 weights
    half_t* w_enc_ih = allocH(768 * 272);
    half_t* w_enc_hh = allocH(768 * 256);
    half_t* w_rel_ih = allocH(768 * 16);
    half_t* w_rel_hh = allocH(768 * 256);
    half_t* w_dec_ih = allocH((size_t)1536 * 512);
    half_t* w_dec_hh = allocH((size_t)1536 * 512);
    half_t* w_read   = allocH((size_t)NH * 257 * 256);   // 1028 x 256
    half_t* w_write  = allocH((size_t)769 * 256);

    // fp32 state & scratch
    float* emb_p  = allocF((size_t)T_PAST * A_TOT * 16);
    float* emb_r  = allocF((size_t)T_PAST * A_TOT * 16);
    float* h_st   = allocF((size_t)A_TOT * C_DIM);
    float* hrel   = allocF((size_t)A_TOT * C_DIM);
    float* r_st   = allocF((size_t)A_TOT * M_DIM);
    float* h_fut  = allocF((size_t)R_DEC * D_DIM);
    float* pres   = allocF((size_t)R_DEC * 2);
    float* mem    = allocF((size_t)S_SCENES * N_SLOT * M_DIM);
    float* x_cat  = allocF((size_t)A_TOT * 272);
    float* gi     = allocF((size_t)A_TOT * 768);
    float* gh     = allocF((size_t)A_TOT * 768);
    float* wp     = allocF((size_t)A_TOT * 769);
    float* rp     = allocF((size_t)A_TOT * 1028);
    float* x_dec  = allocF((size_t)R_DEC * D_DIM);
    float* gid    = allocF((size_t)R_DEC * 1536);
    float* ghd    = allocF((size_t)R_DEC * 1536);

    auto cvt = [&](const float* src, half_t* dst, size_t n) {
        cvt_f16<<<dim3((unsigned)((n + 255) / 256)), dim3(256), 0, stream>>>(src, dst, (int)n);
    };
    auto gate = [&](const float* gi_, const float* gh_, const float* bi, const float* bh,
                    float* hh, int R, int H) {
        size_t n = (size_t)R * H;
        gru_gate<<<dim3((unsigned)((n + 255) / 256)), dim3(256), 0, stream>>>(gi_, gh_, bi, bh, hh, R, H);
    };

    // ---- weight conversion ----
    cvt(enc_wih, w_enc_ih, (size_t)768 * 272);
    cvt(enc_whh, w_enc_hh, (size_t)768 * 256);
    cvt(rel_wih, w_rel_ih, (size_t)768 * 16);
    cvt(rel_whh, w_rel_hh, (size_t)768 * 256);
    cvt(dec_wih, w_dec_ih, (size_t)1536 * 512);
    cvt(dec_whh, w_dec_hh, (size_t)1536 * 512);
    cvt(read_w,  w_read,   (size_t)NH * 257 * 256);
    cvt(write_w, w_write,  (size_t)769 * 256);

    // ---- embeddings (time-major) ----
    {
        unsigned g = (A_TOT * T_PAST + 255) / 256;
        embed_mlp<<<dim3(g), dim3(256), 0, stream>>>(past, ep_w1, ep_b1, ep_w2, ep_b2, emb_p);
        embed_mlp<<<dim3(g), dim3(256), 0, stream>>>(past_rel, er_w1, er_b1, er_w2, er_b2, emb_r);
    }

    // ---- carry init ----
    fill_f32<<<dim3(4096), dim3(256), 0, stream>>>(mem, 1e-6f, (size_t)S_SCENES * N_SLOT * M_DIM);
    bcast_row<<<dim3((A_TOT * C_DIM + 255) / 256), dim3(256), 0, stream>>>(r_st, init_r, A_TOT, M_DIM);
    bcast_row<<<dim3((A_TOT * C_DIM + 255) / 256), dim3(256), 0, stream>>>(h_st, h_bias, A_TOT, C_DIM);
    bcast_row<<<dim3((A_TOT * C_DIM + 255) / 256), dim3(256), 0, stream>>>(hrel, h_bias_rel, A_TOT, C_DIM);
    bcast_row<<<dim3(((unsigned)((size_t)R_DEC * D_DIM + 255) / 256)), dim3(256), 0, stream>>>(h_fut, h_bias_fut, R_DEC, D_DIM);
    init_present<<<dim3((R_DEC * 2 + 255) / 256), dim3(256), 0, stream>>>(pres, past);

    // ---- 20 recurrent steps ----
    for (int t = 0; t < T_PAST + F_FUT; ++t) {
        const bool is_past = (t < T_PAST);
        const float* et  = emb_p + (size_t)(is_past ? t : 0) * A_TOT * 16;
        const float* ert = emb_r + (size_t)(is_past ? t : 0) * A_TOT * 16;

        // rel GRU
        if (is_past) launch_gemm<768, 16>(ert, w_rel_ih, gi, A_TOT, stream);
        else         fill_f32<<<dim3(4096), dim3(256), 0, stream>>>(gi, 0.f, (size_t)A_TOT * 768);
        launch_gemm<768, 256>(hrel, w_rel_hh, gh, A_TOT, stream);
        gate(gi, gh, rel_bih, rel_bhh, hrel, A_TOT, 256);

        // enc GRU on [emb | r]
        concat_emb_r<<<dim3((A_TOT * 272 + 255) / 256), dim3(256), 0, stream>>>(
            et, r_st, x_cat, is_past ? 1 : 0);
        launch_gemm<768, 272>(x_cat, w_enc_ih, gi, A_TOT, stream);
        launch_gemm<768, 256>(h_st, w_enc_hh, gh, A_TOT, stream);
        gate(gi, gh, enc_bih, enc_bhh, h_st, A_TOT, 256);

        // memory write (past steps only)
        if (is_past) {
            launch_gemm<769, 256>(hrel, w_write, wp, A_TOT, stream);
            mem_write<<<dim3(A_TOT), dim3(256), 0, stream>>>(wp, write_b, mem);
        }

        // memory read -> r, x_dec
        launch_gemm<1028, 256>(h_st, w_read, rp, A_TOT, stream);
        mem_read<<<dim3(A_TOT), dim3(256), 0, stream>>>(rp, read_b, mem, h_st, r_st, x_dec);

        // decoder GRU
        launch_gemm<1536, 512>(x_dec, w_dec_ih, gid, R_DEC, stream);
        launch_gemm<1536, 512>(h_fut, w_dec_hh, ghd, R_DEC, stream);
        gate(gid, ghd, dec_bih, dec_bhh, h_fut, R_DEC, 512);

        // output displacement accumulation (future steps)
        if (!is_past) {
            dec_out<<<dim3(R_DEC / 8), dim3(256), 0, stream>>>(
                h_fut, out_w, out_b, pres, (float*)d_out, t - T_PAST);
        }
    }
}
